// TopKNoisyRouter_20091857010895
// MI455X (gfx1250) — compile-verified, matrix operand order checked by reading, not run on hardware
//
#include <hip/hip_runtime.h>
#include <hip/hip_bf16.h>
#include <math.h>

typedef __bf16 bf16_t;
typedef __attribute__((ext_vector_type(16))) __bf16 v16bf;
typedef __attribute__((ext_vector_type(8)))  float  v8f;
typedef __attribute__((ext_vector_type(4)))  float  v4f;

#define NTOK   65536
#define DDIM   1024
#define NEXP   64
#define KCH    32                 // K per wmma_f32_16x16x32_bf16
#define NKCH   (DDIM / KCH)       // 32 k-chunks
#define NTILE  4                  // 4 expert tiles of 16
#define WAVE_M 32                 // tokens per wave (two 16-row M tiles)
#define WAVES  4                  // waves per block
#define BLOCK_M (WAVE_M * WAVES)  // 128 tokens per block
#define LDS_STRIDE 68             // 64 + 4 pad floats -> conflict-free, 16B aligned rows
#define MAT_FRAG_ELEMS 65536      // bf16 elems per packed matrix: 4*32*32*16

// ---------------------------------------------------------------------------
// Kernel 1: convert w_route / w_noise (fp32, [E=64][D=1024]) into bf16 WMMA
// B-fragments in d_ws. Layout per matrix:
//   frag[nt][kc][lane][j]  (j = 0..15 bf16 per lane, 32B contiguous)
// Lane/element -> (n,k) mapping mirrors the CDNA5 16-bit B-matrix layout:
//   n = nt*16 + (lane & 15)
//   k = kc*32 + (lane<16 ? 0 : 8) + (j<8 ? j : j+8)
// ---------------------------------------------------------------------------
__global__ void router_prepack_kernel(const float* __restrict__ w_route,
                                      const float* __restrict__ w_noise,
                                      bf16_t* __restrict__ ws) {
    int id = blockIdx.x * blockDim.x + threadIdx.x;   // 0 .. 131071
    int j    =  id        & 15;
    int lane = (id >> 4)  & 31;
    int kc   = (id >> 9)  & 31;
    int nt   = (id >> 14) & 3;
    int mat  = (id >> 16) & 1;
    const float* w = mat ? w_noise : w_route;
    int n = nt * 16 + (lane & 15);
    int k = kc * 32 + ((lane < 16) ? 0 : 8) + ((j < 8) ? j : j + 8);
    ws[id] = (bf16_t)w[n * DDIM + k];
}

__device__ inline float softplus_f(float v) {
    return fmaxf(v, 0.0f) + log1pf(expf(-fabsf(v)));
}

// ---------------------------------------------------------------------------
// Kernel 2: fused dual-GEMM (route + noise) + noisy logits + top-2 + softmax
// + scatter. One wave handles 32 tokens x 64 experts.
// ---------------------------------------------------------------------------
__global__ __launch_bounds__(WAVES * 32) void
router_main_kernel(const float* __restrict__ x,
                   const float* __restrict__ eps,
                   const bf16_t* __restrict__ ws,
                   float* __restrict__ out,
                   int* __restrict__ idx_out) {
    __shared__ float lds_tile[WAVES][WAVE_M][LDS_STRIDE];

    const int wave = threadIdx.x >> 5;
    const int lane = threadIdx.x & 31;
    const int rowBase = blockIdx.x * BLOCK_M + wave * WAVE_M;

    const v8f vz = {0.f, 0.f, 0.f, 0.f, 0.f, 0.f, 0.f, 0.f};
    v8f accL[2][NTILE];   // route logits, 2 M-tiles x 4 N-tiles
    v8f accN[2][NTILE];   // noise logits
#pragma unroll
    for (int mt = 0; mt < 2; ++mt)
#pragma unroll
        for (int nt = 0; nt < NTILE; ++nt) { accL[mt][nt] = vz; accN[mt][nt] = vz; }

    const int halfsel = (lane < 16) ? 0 : 8;            // A/B K-split per lane half
    const float* xr0 = x + (size_t)(rowBase + (lane & 15)) * DDIM;
    const float* xr1 = xr0 + (size_t)16 * DDIM;

    for (int kc = 0; kc < NKCH; ++kc) {
        const int k0 = kc * KCH + halfsel;              // K 0..7 / 8..15
        const int k1 = k0 + 16;                         // K 16..23 / 24..31

        // --- A fragments (fp32 -> bf16 in registers) ---
        v16bf a[2];
#pragma unroll
        for (int mt = 0; mt < 2; ++mt) {
            const float* px = (mt == 0) ? xr0 : xr1;
            __builtin_prefetch(px + k0 + 2 * KCH, 0, 0);   // global_prefetch next chunks
            v4f f0 = *(const v4f*)(px + k0);
            v4f f1 = *(const v4f*)(px + k0 + 4);
            v4f f2 = *(const v4f*)(px + k1);
            v4f f3 = *(const v4f*)(px + k1 + 4);
#pragma unroll
            for (int i = 0; i < 4; ++i) {
                a[mt][i]      = (__bf16)f0[i];
                a[mt][4 + i]  = (__bf16)f1[i];
                a[mt][8 + i]  = (__bf16)f2[i];
                a[mt][12 + i] = (__bf16)f3[i];
            }
        }

        // --- B fragments (packed bf16, L2/WGP$-resident) + 16 WMMAs ---
#pragma unroll
        for (int nt = 0; nt < NTILE; ++nt) {
            const bf16_t* pb = ws + ((nt * NKCH + kc) * 32 + lane) * 16;
            v16bf bR = *(const v16bf*)pb;
            v16bf bN = *(const v16bf*)(pb + MAT_FRAG_ELEMS);
#pragma unroll
            for (int mt = 0; mt < 2; ++mt) {
                accL[mt][nt] = __builtin_amdgcn_wmma_f32_16x16x32_bf16(
                    false, a[mt], false, bR, (short)0, accL[mt][nt], false, false);
                accN[mt][nt] = __builtin_amdgcn_wmma_f32_16x16x32_bf16(
                    false, a[mt], false, bN, (short)0, accN[mt][nt], false, false);
            }
        }
    }

    // --- Epilogue 1: noisy = logit + eps * softplus(noise) -> LDS tile ---
    // C/D layout: VGPR r, lanes 0-15 -> M=r, N=lane ; lanes 16-31 -> M=r+8, N=lane-16
    float (*tile)[LDS_STRIDE] = lds_tile[wave];
    const int colE = lane & 15;
    const int rAdd = (lane < 16) ? 0 : 8;
#pragma unroll
    for (int mt = 0; mt < 2; ++mt)
#pragma unroll
        for (int nt = 0; nt < NTILE; ++nt)
#pragma unroll
            for (int r = 0; r < 8; ++r) {
                int localRow = mt * 16 + r + rAdd;
                int e = nt * 16 + colE;
                float ep = eps[(size_t)(rowBase + localRow) * NEXP + e];
                float noisy = accL[mt][nt][r] + ep * softplus_f(accN[mt][nt][r]);
                tile[localRow][e] = noisy;
            }

    // --- Epilogue 2: per-token top-2 + softmax + scatter (lane == token row) ---
    const int myRow = lane;
    float v0 = -INFINITY, v1 = -INFINITY;
    int i0 = 0, i1 = 0;
#pragma unroll 8
    for (int j = 0; j < NEXP; ++j) {
        float v = tile[myRow][j];
        if (v > v0) { v1 = v0; i1 = i0; v0 = v; i0 = j; }
        else if (v > v1) { v1 = v; i1 = j; }
    }
    float p0 = 1.0f / (1.0f + expf(v1 - v0));   // softmax over {v0, v1}
    float p1 = 1.0f - p0;

#pragma unroll 8
    for (int j = 0; j < NEXP; ++j) tile[myRow][j] = 0.0f;
    tile[myRow][i0] = p0;
    tile[myRow][i1] = p1;

    idx_out[(size_t)(rowBase + myRow) * 2 + 0] = i0;
    idx_out[(size_t)(rowBase + myRow) * 2 + 1] = i1;

    // --- Epilogue 3: coalesced b128 copy LDS tile -> global [32][64] ---
    float* outBase = out + (size_t)rowBase * NEXP;
#pragma unroll
    for (int i = 0; i < 16; ++i) {
        int flat = i * 32 + lane;              // float4 index 0..511
        int row  = flat >> 4;
        int col  = (flat & 15) * 4;
        v4f val = *(const v4f*)&tile[row][col];
        *(v4f*)(outBase + (size_t)flat * 4) = val;
    }
}

// ---------------------------------------------------------------------------
// Host launcher. Inputs: x[N*D] f32, w_route[E*D] f32, w_noise[E*D] f32,
// eps[N*E] f32. Output: router_output[N*E] f32 followed by indices[N*2] i32.
// ---------------------------------------------------------------------------
extern "C" void kernel_launch(void* const* d_in, const int* in_sizes, int n_in,
                              void* d_out, int out_size, void* d_ws, size_t ws_size,
                              hipStream_t stream) {
    const float* x       = (const float*)d_in[0];
    const float* w_route = (const float*)d_in[1];
    const float* w_noise = (const float*)d_in[2];
    const float* eps     = (const float*)d_in[3];
    float* out = (float*)d_out;
    int*   idx = (int*)(out + (size_t)NTOK * NEXP);
    bf16_t* ws = (bf16_t*)d_ws;   // 2 * 65536 bf16 = 256 KB

    // Pack both weight matrices into bf16 WMMA B-fragments (deterministic,
    // rerun every call; same stream => ordered before the main kernel).
    router_prepack_kernel<<<(2 * MAT_FRAG_ELEMS) / 256, 256, 0, stream>>>(w_route, w_noise, ws);

    // 512 blocks x 128 threads (4 waves); each wave: 32 tokens x 64 experts.
    router_main_kernel<<<NTOK / BLOCK_M, WAVES * 32, 0, stream>>>(x, eps, ws, out, idx);
}
